// Attribution_47536698032923
// MI455X (gfx1250) — matrix operand order repaired and checked by reading, not compiled, not run on hardware
//
#include <hip/hip_runtime.h>

// ---------------------------------------------------------------------------
// MI455X (gfx1250):
//   K1: h16   = f16-WMMA GEMM (x[nchw] * W1^T + b1)            -> ws f16 pixel-major
//   K2: gate  = sigmoid(relu(f16-WMMA GEMM (h * W2^T + b2)))   -> ws fp32 pixel-major
//   K3: 5x5 neighbor corr + softmax + sample, fused gate mult  -> ws f16 (attr)
//   K4: out   = f16-WMMA GEMM (attr * Wout^T + bout)           -> d_out [n,cin,h,w]
// GEMMs: double-buffered LDS staging; f16 activations staged via async
// global->LDS copies (GLOBAL_LOAD_ASYNC_TO_LDS_B128 + s_wait_asynccnt).
// ---------------------------------------------------------------------------

typedef __attribute__((ext_vector_type(16))) _Float16     v16h;
typedef __attribute__((ext_vector_type(8)))  _Float16     h8v;
typedef __attribute__((ext_vector_type(4)))  _Float16     h4v;
typedef __attribute__((ext_vector_type(8)))  float        v8f;
typedef __attribute__((ext_vector_type(4)))  unsigned int uint4v;
typedef __attribute__((ext_vector_type(4)))  int          int4v;

union Frag16 {
    uint4v q[2];
    v16h   h;
};

// problem dims
constexpr int NB   = 8;
constexpr int CIN  = 256;
constexpr int CH   = 128;
constexpr int HW   = 64 * 64;        // 4096
constexpr int NPIX = NB * HW;        // 32768
constexpr int BM   = 128;            // pixels per block (8 waves x 16 rows)
constexpr int APAD = 40;             // f16 per staged A row (80B: 16B aligned, bank-spread)
constexpr int WPAD = 40;             // f16 per staged W row

// ---- gfx1250 async global->LDS copy (guarded; falls back to sync copy) ----
#if defined(__has_builtin)
#  if __has_builtin(__builtin_amdgcn_global_load_async_to_lds_b128)
#    define USE_ASYNC_LDS 1
#  endif
#endif
#ifndef USE_ASYNC_LDS
#  define USE_ASYNC_LDS 0
#endif

#define GLOBAL_AS __attribute__((address_space(1)))
#define LDS_AS    __attribute__((address_space(3)))

__device__ __forceinline__ void copy16_g2l(void* lds, const void* g)
{
#if USE_ASYNC_LDS
    __builtin_amdgcn_global_load_async_to_lds_b128(
        (GLOBAL_AS int4v*)(g),
        (LDS_AS int4v*)(lds), 0, 0);
#else
    *(uint4v*)lds = *(const uint4v*)g;
#endif
}

__device__ __forceinline__ void async_wait()
{
#if USE_ASYNC_LDS
#  if __has_builtin(__builtin_amdgcn_s_wait_asynccnt)
    __builtin_amdgcn_s_wait_asynccnt(0);
#  else
    asm volatile("s_wait_asynccnt 0" ::: "memory");
#  endif
#endif
}

// ---------------------------------------------------------------------------
// out[p,o] = sum_c A[p,c] * Wt[o,c] + bias[o]
//  AMODE 0: A is x fp32 (n,c,h,w); AMODE 1: A is f16 pixel-major (raw copy)
//  EPI 0: bias -> f16 pixel-major | EPI 1: bias,relu,sigmoid -> fp32 pixel-major
//  EPI 2: bias -> fp32 channel-major (n,o,y,x)
// ---------------------------------------------------------------------------
template<int KTOT, int OTOT, int AMODE, int EPI>
__global__ __launch_bounds__(256) void gemm_conv1x1(
    const void* __restrict__ Asrc, const float* __restrict__ Wt,
    const float* __restrict__ bias, void* __restrict__ outv)
{
    __shared__ __align__(16) _Float16 sA[2 * BM * APAD];
    __shared__ __align__(16) _Float16 sW[2 * OTOT * WPAD];

    const int tid  = threadIdx.x;
    const int lane = tid & 31;
    const int wv   = tid >> 5;
    const int p0   = blockIdx.x * BM;
    const int nimg = p0 >> 12;           // p0 / HW
    const int yx0  = p0 & (HW - 1);

    constexpr int NT = OTOT / 16;
    constexpr int NK = KTOT / 32;
    v8f acc[NT] = {};

    const int half = lane >> 4;
    const int nloc = lane & 15;

    // ---- staging helpers ----
    auto stageA = [&](int buf, int k0) {
        _Float16* sAb = sA + buf * (BM * APAD);
        if (AMODE == 0) {
            const float* A = (const float*)Asrc;
            const size_t base = (size_t)nimg * KTOT * HW + yx0;
            for (int i = tid; i < BM * 32 / 4; i += 256) {   // unit: 4 px, 1 ch
                const int c = i >> 5, u = i & 31;
                const float* src = A + base + (size_t)(k0 + c) * HW + u * 4;
                if (k0 + 32 < KTOT) __builtin_prefetch(src + 32 * HW, 0, 0);
                const float4 v = *(const float4*)src;
                _Float16* d = sAb + (u * 4) * APAD + c;
                d[0 * APAD] = (_Float16)v.x;
                d[1 * APAD] = (_Float16)v.y;
                d[2 * APAD] = (_Float16)v.z;
                d[3 * APAD] = (_Float16)v.w;
            }
        } else {
            const _Float16* A16 = (const _Float16*)Asrc;
            for (int i = tid; i < BM * 4; i += 256) {        // unit: 16B chunk
                const int pix = i >> 2, j = i & 3;
                copy16_g2l(sAb + pix * APAD + j * 8,
                           A16 + (size_t)(p0 + pix) * KTOT + k0 + j * 8);
            }
        }
    };
    auto stageW = [&](int buf, int k0) {
        _Float16* sWb = sW + buf * (OTOT * WPAD);
        for (int i = tid; i < OTOT * 8; i += 256) {          // unit: 1 row, 4 ch
            const int o = i >> 3, u = i & 7;
            const float4 v = *(const float4*)(Wt + (size_t)o * KTOT + k0 + u * 4);
            h4v hv = { (_Float16)v.x, (_Float16)v.y, (_Float16)v.z, (_Float16)v.w };
            *(h4v*)(sWb + o * WPAD + u * 4) = hv;
        }
    };

    stageA(0, 0);
    stageW(0, 0);
    async_wait();
    __syncthreads();

    for (int kc = 0; kc < NK; ++kc) {
        const int buf = kc & 1;
        if (kc + 1 < NK) {                   // overlap next-chunk staging with WMMA
            stageA(buf ^ 1, (kc + 1) * 32);
            stageW(buf ^ 1, (kc + 1) * 32);
        }

        const _Float16* sAb = sA + buf * (BM * APAD);
        const _Float16* sWb = sW + buf * (OTOT * WPAD);

        // A fragment (16-bit A 16x32, ISA 7.12.2):
        //  lanes 0-15 : row M=lane,  v0-3 = K 0..7,  v4-7 = K 16..23
        //  lanes 16-31: row M=l-16,  v0-3 = K 8..15, v4-7 = K 24..31
        Frag16 fa;
        const _Float16* ar = sAb + ((wv << 4) + nloc) * APAD;
        fa.q[0] = *(const uint4v*)(ar + (half << 3));
        fa.q[1] = *(const uint4v*)(ar + 16 + (half << 3));

#pragma unroll
        for (int t = 0; t < NT; ++t) {
            // B fragment (K=32 x N=16): lane holds column N=lane&15;
            //  lanes 0-15 K 0..7 / 16..23, lanes 16-31 K 8..15 / 24..31 (matches A)
            Frag16 fb;
            const _Float16* br = sWb + ((t << 4) + nloc) * WPAD + (half << 3);
            fb.q[0] = *(const uint4v*)(br);
            fb.q[1] = *(const uint4v*)(br + 16);

            acc[t] = __builtin_amdgcn_wmma_f32_16x16x32_f16(
                false, fa.h, false, fb.h, (short)0, acc[t], false, false);
        }
        async_wait();
        __syncthreads();
    }

    // ---- epilogue: D layout lane -> (M = v + 8*half, N = lane&15) ----
#pragma unroll
    for (int t = 0; t < NT; ++t) {
        const int o  = (t << 4) + nloc;
        const float bo = bias[o];
#pragma unroll
        for (int v = 0; v < 8; ++v) {
            const int p = p0 + (wv << 4) + v + (half << 3);
            float val = acc[t][v] + bo;
            if (EPI == 0) {
                ((_Float16*)outv)[(size_t)p * OTOT + o] = (_Float16)val;
            } else if (EPI == 1) {
                float r = fmaxf(val, 0.f);
                ((float*)outv)[(size_t)p * OTOT + o] = 1.f / (1.f + __expf(-r));
            } else {
                ((float*)outv)[(size_t)(p >> 12) * OTOT * HW + (size_t)o * HW +
                               (p & (HW - 1))] = val;
            }
        }
    }
}

// ---------------------------------------------------------------------------
// K3: 25-neighbor correlation + softmax + sample, fused with gate multiply.
// 4 lanes per pixel (32 ch each); butterfly shfl_xor reduce within the group.
// h is f16 (halved traffic, L2-resident); softmax/gate math in fp32.
// ---------------------------------------------------------------------------
__global__ __launch_bounds__(256) void neighbor_attn(
    const _Float16* __restrict__ H16, const float* __restrict__ Gate,
    _Float16* __restrict__ Attr)
{
    const int tid = threadIdx.x;
    const int q   = tid & 3;
    const int p   = blockIdx.x * 64 + (tid >> 2);
    const int yx  = p & (HW - 1);
    const int y   = yx >> 6, x = yx & 63;

    const h8v* cp = (const h8v*)(H16 + (size_t)p * CH + q * 32);
    float cf[32];
#pragma unroll
    for (int j = 0; j < 4; ++j) {
        h8v v = cp[j];
#pragma unroll
        for (int e = 0; e < 8; ++e) cf[j * 8 + e] = (float)v[e];
    }

    float corr[25];
#pragma unroll
    for (int k = 0; k < 25; ++k) {
        const int dy = k / 5 - 2, dx = k % 5 - 2;
        const int yy = y + dy, xx = x + dx;
        float s = 0.f;
        if ((unsigned)yy < 64u && (unsigned)xx < 64u) {
            const h8v* nb = (const h8v*)(H16 + (size_t)(p + dy * 64 + dx) * CH + q * 32);
#pragma unroll
            for (int j = 0; j < 4; ++j) {
                h8v b = nb[j];
#pragma unroll
                for (int e = 0; e < 8; ++e) s = fmaf(cf[j * 8 + e], (float)b[e], s);
            }
        }
        s += __shfl_xor(s, 1, 32);
        s += __shfl_xor(s, 2, 32);
        corr[k] = s * 0.08838834764831845f;   // 1/sqrt(128)
    }

    // softmax over 25 neighbors (zero-padded OOB contribute corr = 0)
    float m = corr[0];
#pragma unroll
    for (int k = 1; k < 25; ++k) m = fmaxf(m, corr[k]);
    float tot = 0.f;
#pragma unroll
    for (int k = 0; k < 25; ++k) { corr[k] = __expf(corr[k] - m); tot += corr[k]; }
    const float inv = 1.f / tot;

    float accs[32] = {};
#pragma unroll
    for (int k = 0; k < 25; ++k) {
        const int dy = k / 5 - 2, dx = k % 5 - 2;
        const int yy = y + dy, xx = x + dx;
        if ((unsigned)yy < 64u && (unsigned)xx < 64u) {
            const float a = corr[k] * inv;
            const h8v* nb = (const h8v*)(H16 + (size_t)(p + dy * 64 + dx) * CH + q * 32);
#pragma unroll
            for (int j = 0; j < 4; ++j) {
                h8v b = nb[j];
#pragma unroll
                for (int e = 0; e < 8; ++e)
                    accs[j * 8 + e] = fmaf(a, (float)b[e], accs[j * 8 + e]);
            }
        }
    }

    const float4* gp = (const float4*)(Gate + (size_t)p * CH + q * 32);
    h8v* op = (h8v*)(Attr + (size_t)p * CH + q * 32);
#pragma unroll
    for (int j = 0; j < 4; ++j) {
        const float4 g0 = gp[2 * j], g1 = gp[2 * j + 1];
        h8v o;
        o[0] = (_Float16)(g0.x * accs[j * 8 + 0]);
        o[1] = (_Float16)(g0.y * accs[j * 8 + 1]);
        o[2] = (_Float16)(g0.z * accs[j * 8 + 2]);
        o[3] = (_Float16)(g0.w * accs[j * 8 + 3]);
        o[4] = (_Float16)(g1.x * accs[j * 8 + 4]);
        o[5] = (_Float16)(g1.y * accs[j * 8 + 5]);
        o[6] = (_Float16)(g1.z * accs[j * 8 + 6]);
        o[7] = (_Float16)(g1.w * accs[j * 8 + 7]);
        op[j] = o;
    }
}

// ---------------------------------------------------------------------------
extern "C" void kernel_launch(void* const* d_in, const int* in_sizes, int n_in,
                              void* d_out, int out_size, void* d_ws, size_t ws_size,
                              hipStream_t stream)
{
    (void)in_sizes; (void)n_in; (void)out_size; (void)ws_size;

    const float* x    = (const float*)d_in[0];
    const float* W1   = (const float*)d_in[1];
    const float* b1   = (const float*)d_in[2];
    const float* W2   = (const float*)d_in[3];
    const float* b2   = (const float*)d_in[4];
    const float* Wout = (const float*)d_in[5];
    const float* bout = (const float*)d_in[6];
    float* out = (float*)d_out;

    // ws: h16 (8MB) | attr16 (8MB) | gate fp32 (16MB)
    _Float16* h16  = (_Float16*)d_ws;
    _Float16* a16  = h16 + (size_t)NPIX * CH;
    float*    gate = (float*)(a16 + (size_t)NPIX * CH);

    gemm_conv1x1<CIN, CH, 0, 0><<<NPIX / BM, 256, 0, stream>>>(x,   W1,   b1,   h16);
    gemm_conv1x1<CH,  CH, 1, 1><<<NPIX / BM, 256, 0, stream>>>(h16, W2,   b2,   gate);
    neighbor_attn              <<<NPIX / 64, 256, 0, stream>>>(h16, gate, a16);
    gemm_conv1x1<CH, CIN, 1, 2><<<NPIX / BM, 256, 0, stream>>>(a16, Wout, bout, out);
}
